// EwaldBlock_13142599926313
// MI455X (gfx1250) — matrix operand order repaired and checked by reading, not compiled, not run on hardware
//
#include <hip/hip_runtime.h>
#include <hip/hip_bf16.h>

#define EDIM 256
#define KDIM 64
#define BDIM 16
#define WPITCH 264   // bf16 elements per LDS weight row: 528B = 132 dwords -> rows hit distinct banks, 16B aligned

typedef __bf16 v16bf __attribute__((ext_vector_type(16)));
typedef float  v8f   __attribute__((ext_vector_type(8)));

#define INV_SQRT2 0.70710678118654752f
#define SILU_SC   (1.0f / 0.6f)

__device__ __forceinline__ float ssilu(float v) {
  // scaled SiLU: x * sigmoid(x) / 0.6
  return (v * SILU_SC) / (1.0f + __expf(-v));
}

// Async global->LDS copy of 16 bytes per lane (CDNA5, tracked by ASYNCcnt).
__device__ __forceinline__ void async_copy_b128(const void* gsrc, void* ldst) {
  const unsigned lds_off = (unsigned)(size_t)ldst;            // low 32 bits = LDS offset
  const unsigned long long ga = (unsigned long long)(size_t)gsrc;
  asm volatile("global_load_async_to_lds_b128 %0, %1, off"
               :: "v"(lds_off), "v"(ga) : "memory");
}
__device__ __forceinline__ void async_wait0() {
  asm volatile("s_wait_asynccnt 0x0" ::: "memory");
}

// ---------------------------------------------------------------------------
// Convert the 9 dense-weight matrices (W_pre[2], W_e0, W_res[3][2]) to bf16.
// ---------------------------------------------------------------------------
__global__ __launch_bounds__(256) void k_cvt(const float* __restrict__ wpre,
                                             const float* __restrict__ we0,
                                             const float* __restrict__ wres,
                                             __bf16* __restrict__ outw) {
  const int idx = blockIdx.x * 256 + threadIdx.x;   // 9 * 65536 total
  const int m = idx >> 16;
  const int r = idx & 65535;
  float v;
  if (m < 2)       v = wpre[m * 65536 + r];
  else if (m == 2) v = we0[r];
  else             v = wres[(m - 3) * 65536 + r];
  outw[idx] = (__bf16)v;
}

// ---------------------------------------------------------------------------
// dot[n,k] = k[batch_seg[n],k,:] . x[n,:]; c = cos(dot), s = sin(dot)
// ---------------------------------------------------------------------------
__global__ __launch_bounds__(256) void k_dot(const float* __restrict__ x,
                                             const float* __restrict__ kvec,
                                             const int* __restrict__ bseg,
                                             float* __restrict__ dotOut,
                                             float* __restrict__ cOut,
                                             float* __restrict__ sOut,
                                             float* __restrict__ scalarOut) {
  const int idx = blockIdx.x * 256 + threadIdx.x;   // N*K total
  const int n  = idx >> 6;
  const int kk = idx & 63;
  const int b  = bseg[n];
  const float* kp = kvec + (size_t)(b * KDIM + kk) * 3;
  const float* xp = x + (size_t)n * 3;
  const float d = kp[0] * xp[0] + kp[1] * xp[1] + kp[2] * xp[2];
  dotOut[idx] = d;
  float sv, cv;
  __sincosf(d, &sv, &cv);
  cOut[idx] = cv;
  sOut[idx] = sv;
  if (idx == 0) scalarOut[0] = 1.0f;
}

// ---------------------------------------------------------------------------
// kfilter[k,e] = sum_d W_up[e,d] * W_down[d,k]   (D = 8)
// ---------------------------------------------------------------------------
__global__ __launch_bounds__(256) void k_kfilt(const float* __restrict__ wdown,
                                               const float* __restrict__ wup,
                                               float* __restrict__ kf) {
  const int idx = blockIdx.x * 256 + threadIdx.x;   // K*E = 16384
  const int k = idx >> 8;
  const int e = idx & 255;
  float acc = 0.f;
  #pragma unroll
  for (int d = 0; d < 8; ++d) acc += wup[e * 8 + d] * wdown[d * KDIM + k];
  kf[k * EDIM + e] = acc;
}

// ---------------------------------------------------------------------------
// Per-batch segment starts (batch_seg is sorted): seg[b] = lower_bound(b)
// ---------------------------------------------------------------------------
__global__ void k_seg(const int* __restrict__ bseg, int* __restrict__ seg, int N) {
  const int b = threadIdx.x;
  if (b > BDIM) return;
  int lo = 0, hi = N;
  while (lo < hi) {
    const int mid = (lo + hi) >> 1;
    if (bseg[mid] < b) lo = mid + 1; else hi = mid;
  }
  seg[b] = lo;
}

// ---------------------------------------------------------------------------
// WMMA GEMM: Y[n,e] = epilogue( sum_j X[n,j] * W[e,j] )
//   epilogue: residual ? (ssilu(acc) + R) * 1/sqrt(2) : ssilu(acc)
// Block = 256 threads (8 waves). Block tile = 128(M) x 64(N). K = E = 256.
// Weight slab (64 e-rows x 256 j, bf16) staged in LDS via async global->LDS.
// A fragments (8 K-steps) preloaded to registers to avoid WMMA->VALU hazards.
// Each wave: 8 K-steps x 4 N-tiles of v_wmma_f32_16x16x32_bf16.
// ---------------------------------------------------------------------------
__global__ __launch_bounds__(256) void k_gemm(const float* __restrict__ X,
                                              const __bf16* __restrict__ Wg,
                                              const float* __restrict__ R,
                                              float* __restrict__ Y,
                                              int residual) {
  __shared__ __bf16 Wsm[64 * WPITCH];
  const int t    = threadIdx.x;
  const int wave = t >> 5;
  const int lane = t & 31;
  const int m0   = blockIdx.x * 128 + wave * 16;
  const int e0   = blockIdx.y * 64;

  { // async-stage the 64x256 bf16 weight tile into LDS (8 x 16B per thread)
    const int r = t >> 2;
    const int c = (t & 3) * 64;
    const __bf16* src = Wg + (size_t)(e0 + r) * EDIM + c;
    __bf16* dst = Wsm + r * WPITCH + c;
    #pragma unroll
    for (int i = 0; i < 8; ++i) async_copy_b128(src + i * 8, dst + i * 8);
    async_wait0();          // drain this wave's ASYNCcnt before the barrier
  }
  __syncthreads();

  const bool lowHalf = (lane < 16);
  const int kbA  = lowHalf ? 0 : 16;   // A: low lanes K0-15, high lanes K16-31
  const int cbB  = lowHalf ? 0 : 16;   // B: low lanes K0-15, high lanes K16-31
  const int ecol = lane & 15;
  const float* xrow = X + (size_t)(m0 + ecol) * EDIM;  // lane&15 selects the M row

  // Preload + convert all 8 A fragments (16 bf16 each) into registers.
  v16bf afr[8];
  #pragma unroll
  for (int ks = 0; ks < 8; ++ks) {
    const int k0 = ks * 32;
    #pragma unroll
    for (int i = 0; i < 8; ++i) afr[ks][i]     = (__bf16)xrow[k0 + kbA + i];
    #pragma unroll
    for (int i = 0; i < 8; ++i) afr[ks][8 + i] = (__bf16)xrow[k0 + kbA + 8 + i];
  }

  v8f acc[4] = {};
  #pragma unroll
  for (int ks = 0; ks < 8; ++ks) {
    const int k0 = ks * 32;
    // issue all B-fragment LDS loads for this K-step first, then 4 WMMAs
    v16bf b[4];
    #pragma unroll
    for (int nt = 0; nt < 4; ++nt) {
      const __bf16* wp = Wsm + (nt * 16 + ecol) * WPITCH + k0 + cbB;
      #pragma unroll
      for (int i = 0; i < 16; ++i) b[nt][i] = wp[i];
    }
    #pragma unroll
    for (int nt = 0; nt < 4; ++nt) {
      acc[nt] = __builtin_amdgcn_wmma_f32_16x16x32_bf16(
          false, afr[ks], false, b[nt], (short)0, acc[nt], false, false);
    }
  }

  // D layout: lane -> col (lane&15); VGPR r -> row r (lanes 0-15) / row 8+r (lanes 16-31)
  #pragma unroll
  for (int nt = 0; nt < 4; ++nt) {
    #pragma unroll
    for (int r = 0; r < 8; ++r) {
      const int m = m0 + (lowHalf ? r : 8 + r);
      const int e = e0 + nt * 16 + ecol;
      float v = ssilu(acc[nt][r]);
      if (residual) v = (v + R[(size_t)m * EDIM + e]) * INV_SQRT2;
      Y[(size_t)m * EDIM + e] = v;
    }
  }
}

// ---------------------------------------------------------------------------
// Structure factors: sf_{r,i}[b,k,e] = sum_{n in seg b} {c,s}[n,k] * hres[n,e]
// Block = (b, e-chunk of 64). Thread = (k, 16 e's). LDS-tiled over 32 atoms.
// ---------------------------------------------------------------------------
__global__ __launch_bounds__(256) void k_sf(const float* __restrict__ hres,
                                            const float* __restrict__ cb,
                                            const float* __restrict__ sb,
                                            const int* __restrict__ seg,
                                            float* __restrict__ sfr,
                                            float* __restrict__ sfi) {
  __shared__ float hh[32][68];
  __shared__ float cc[32][64];
  __shared__ float ss[32][64];
  const int b  = blockIdx.x;
  const int ec = blockIdx.y;          // e-chunk (0..3)
  const int t  = threadIdx.x;
  const int k  = t >> 2;
  const int eq = (t & 3) * 16;
  const int nbeg = seg[b], nend = seg[b + 1];

  float ar[16], ai[16];
  #pragma unroll
  for (int e = 0; e < 16; ++e) { ar[e] = 0.f; ai[e] = 0.f; }

  const int li = t >> 3;
  const int lo = (t & 7) * 8;
  for (int n0 = nbeg; n0 < nend; n0 += 32) {
    __syncthreads();
    const int n = n0 + li;
    if (n < nend) {
      const float* hp = hres + (size_t)n * EDIM + ec * 64 + lo;
      const float* cp = cb + (size_t)n * KDIM + lo;
      const float* sp = sb + (size_t)n * KDIM + lo;
      #pragma unroll
      for (int j = 0; j < 8; ++j) { hh[li][lo + j] = hp[j]; cc[li][lo + j] = cp[j]; ss[li][lo + j] = sp[j]; }
    } else {
      #pragma unroll
      for (int j = 0; j < 8; ++j) { hh[li][lo + j] = 0.f; cc[li][lo + j] = 0.f; ss[li][lo + j] = 0.f; }
    }
    __syncthreads();
    const int lim = (nend - n0 < 32) ? (nend - n0) : 32;
    for (int i = 0; i < lim; ++i) {
      const float cv = cc[i][k];
      const float sv = ss[i][k];
      #pragma unroll
      for (int e = 0; e < 16; ++e) {
        const float hv = hh[i][eq + e];
        ar[e] = __builtin_fmaf(cv, hv, ar[e]);
        ai[e] = __builtin_fmaf(sv, hv, ai[e]);
      }
    }
  }
  float* pr = sfr + (size_t)(b * KDIM + k) * EDIM + ec * 64 + eq;
  float* pi = sfi + (size_t)(b * KDIM + k) * EDIM + ec * 64 + eq;
  #pragma unroll
  for (int e = 0; e < 16; ++e) { pr[e] = ar[e]; pi[e] = ai[e]; }
}

// ---------------------------------------------------------------------------
// sf *= kfilter[k,e] * 0.01  (broadcast over b)
// ---------------------------------------------------------------------------
__global__ __launch_bounds__(256) void k_fmul(float* __restrict__ sfr,
                                              float* __restrict__ sfi,
                                              const float* __restrict__ kf) {
  const int idx = blockIdx.x * 256 + threadIdx.x;   // B*K*E
  const float f = kf[idx & (KDIM * EDIM - 1)] * 0.01f;
  sfr[idx] *= f;
  sfi[idx] *= f;
}

// ---------------------------------------------------------------------------
// h0[n,e] = sum_k c[n,k]*sfr[b,k,e] + s[n,k]*sfi[b,k,e]   (b = batch_seg[n])
// Block per atom; thread per channel e; c/s row cached in LDS.
// ---------------------------------------------------------------------------
__global__ __launch_bounds__(256) void k_gather(const float* __restrict__ cb,
                                                const float* __restrict__ sb,
                                                const float* __restrict__ sfr,
                                                const float* __restrict__ sfi,
                                                const int* __restrict__ bseg,
                                                float* __restrict__ out) {
  __shared__ float cl[64];
  __shared__ float sl[64];
  const int n = blockIdx.x;
  const int t = threadIdx.x;
  if (t < 64)       cl[t] = cb[(size_t)n * KDIM + t];
  else if (t < 128) sl[t - 64] = sb[(size_t)n * KDIM + (t - 64)];
  __syncthreads();
  const int b = bseg[n];
  const float* fr = sfr + (size_t)b * KDIM * EDIM + t;
  const float* fi = sfi + (size_t)b * KDIM * EDIM + t;
  float acc = 0.f;
  #pragma unroll 4
  for (int k = 0; k < KDIM; ++k) {
    acc = __builtin_fmaf(cl[k], fr[(size_t)k * EDIM], acc);
    acc = __builtin_fmaf(sl[k], fi[(size_t)k * EDIM], acc);
  }
  out[(size_t)n * EDIM + t] = acc;
}

// ---------------------------------------------------------------------------
extern "C" void kernel_launch(void* const* d_in, const int* in_sizes, int n_in,
                              void* d_out, int out_size, void* d_ws, size_t ws_size,
                              hipStream_t stream) {
  (void)n_in; (void)out_size; (void)ws_size;
  const float* h     = (const float*)d_in[0];
  const float* x     = (const float*)d_in[1];
  const float* kvec  = (const float*)d_in[2];
  const int*   bseg  = (const int*)d_in[4];
  const float* Wdown = (const float*)d_in[5];
  const float* Wup   = (const float*)d_in[6];
  const float* Wpre  = (const float*)d_in[7];
  const float* We0   = (const float*)d_in[8];
  const float* Wres  = (const float*)d_in[9];
  const int N = in_sizes[0] / EDIM;   // 8192

  // workspace layout (~23.3 MB)
  char* w = (char*)d_ws;
  float* buf0 = (float*)w; w += (size_t)N * EDIM * 4;
  float* buf1 = (float*)w; w += (size_t)N * EDIM * 4;
  float* cbuf = (float*)w; w += (size_t)N * KDIM * 4;
  float* sbuf = (float*)w; w += (size_t)N * KDIM * 4;
  float* sfr  = (float*)w; w += (size_t)BDIM * KDIM * EDIM * 4;
  float* sfi  = (float*)w; w += (size_t)BDIM * KDIM * EDIM * 4;
  float* kf   = (float*)w; w += (size_t)KDIM * EDIM * 4;
  __bf16* wbf = (__bf16*)w; w += (size_t)9 * EDIM * EDIM * 2;
  int*   seg  = (int*)w;

  float* outH   = (float*)d_out;                 // [N,E]
  float* outDot = outH + (size_t)N * EDIM;       // [N,K]
  float* outSc  = outDot + (size_t)N * KDIM;     // scalar

  const dim3 blk(256);
  const dim3 ggrid(N / 128, 4);

  // independent prologue work
  k_cvt  <<<dim3(9 * EDIM * EDIM / 256), blk, 0, stream>>>(Wpre, We0, Wres, wbf);
  k_dot  <<<dim3((N * KDIM) / 256),      blk, 0, stream>>>(x, kvec, bseg, outDot, cbuf, sbuf, outSc);
  k_kfilt<<<dim3(KDIM * EDIM / 256),     blk, 0, stream>>>(Wdown, Wup, kf);
  k_seg  <<<dim3(1), dim3(32),           0, stream>>>(bseg, seg, N);

  // hres = (h + ssilu(ssilu(h@Wpre0.T)@Wpre1.T)) / sqrt(2)
  k_gemm<<<ggrid, blk, 0, stream>>>(h,    wbf + 0 * 65536, nullptr, buf0, 0);
  k_gemm<<<ggrid, blk, 0, stream>>>(buf0, wbf + 1 * 65536, h,       buf1, 1);

  // structure factors, filter, gather-back
  k_sf    <<<dim3(BDIM, 4),                     blk, 0, stream>>>(buf1, cbuf, sbuf, seg, sfr, sfi);
  k_fmul  <<<dim3(BDIM * KDIM * EDIM / 256),    blk, 0, stream>>>(sfr, sfi, kf);
  k_gather<<<dim3(N),                           blk, 0, stream>>>(cbuf, sbuf, sfr, sfi, bseg, buf0);

  // Ewald MLP: Dense(W_e0) + 3 residual layers
  k_gemm<<<ggrid, blk, 0, stream>>>(buf0, wbf + 2 * 65536, nullptr, outH, 0);
  k_gemm<<<ggrid, blk, 0, stream>>>(outH, wbf + 3 * 65536, nullptr, buf0, 0);
  k_gemm<<<ggrid, blk, 0, stream>>>(buf0, wbf + 4 * 65536, outH,    buf1, 1);
  k_gemm<<<ggrid, blk, 0, stream>>>(buf1, wbf + 5 * 65536, nullptr, buf0, 0);
  k_gemm<<<ggrid, blk, 0, stream>>>(buf0, wbf + 6 * 65536, buf1,    outH, 1);
  k_gemm<<<ggrid, blk, 0, stream>>>(outH, wbf + 7 * 65536, nullptr, buf0, 0);
  k_gemm<<<ggrid, blk, 0, stream>>>(buf0, wbf + 8 * 65536, outH,    outH, 1);  // elementwise R/Y alias is safe
}
